// VMamba4DBlock_58892591563290
// MI455X (gfx1250) — matrix operand order repaired and checked by reading, not compiled, not run on hardware
//
#include <hip/hip_runtime.h>
#include <hip/hip_bf16.h>

// ---------------------------------------------------------------------------
// VMamba SS2D block for gfx1250 (MI455X).
// bf16 WMMA GEMMs, async global->LDS tile feed (ASYNCcnt), double-buffered LDS.
// Shapes fixed by the reference: B=8, H=W=64, N=4096, Dm=384, Di=768.
// ---------------------------------------------------------------------------

#define DM   384
#define DI   768
#define BB   8
#define HH   64
#define WW   64
#define NN   (HH*WW)          // 4096
#define MROWS (BB*NN)         // 32768

typedef __attribute__((ext_vector_type(16))) __bf16 v16bf;
typedef __attribute__((ext_vector_type(8)))  float  v8f;

union FragBF { v16bf v; uint4 q[2]; };

__device__ __forceinline__ unsigned short f2bf(float f) {
    union { float f; unsigned u; } x; x.f = f;
    unsigned u = x.u;
    u += 0x7FFFu + ((u >> 16) & 1u);     // round-to-nearest-even
    return (unsigned short)(u >> 16);
}

__device__ __forceinline__ float wave_sum(float v) {
    #pragma unroll
    for (int o = 16; o > 0; o >>= 1) v += __shfl_xor(v, o, 32);
    return v;
}

__device__ __forceinline__ float sigmoidf_(float x) { return 1.0f / (1.0f + expf(-x)); }

// gfx1250 async global->LDS copy, 16B per lane, tracked with ASYNCcnt.
__device__ __forceinline__ void async_ld16(unsigned lds_addr, const void* gptr) {
    asm volatile("global_load_async_to_lds_b128 %0, %1, off"
                 :: "v"(lds_addr), "v"((unsigned long long)(uintptr_t)gptr)
                 : "memory");
}
__device__ __forceinline__ void wait_async0() {
    asm volatile("s_wait_asynccnt 0x0" ::: "memory");
}

// ---------------------------------------------------------------------------
// f32 -> bf16 conversion (weights)
// ---------------------------------------------------------------------------
__global__ void cvt_bf16_kernel(const float* __restrict__ src,
                                unsigned short* __restrict__ dst, int n) {
    int i = blockIdx.x * blockDim.x + threadIdx.x;
    int stride = gridDim.x * blockDim.x;
    for (; i < n; i += stride) dst[i] = f2bf(src[i]);
}

// ---------------------------------------------------------------------------
// LayerNorm over D_MODEL=384, one wave per row, bf16 output
// ---------------------------------------------------------------------------
__global__ __launch_bounds__(256)
void ln_in_kernel(const float* __restrict__ x,
                  const float* __restrict__ w, const float* __restrict__ b,
                  unsigned short* __restrict__ out) {
    const int lane = threadIdx.x & 31, wid = threadIdx.x >> 5;
    const int row = blockIdx.x * 8 + wid;
    const float* xr = x + (size_t)row * DM;
    float vals[12], s = 0.f, ss = 0.f;
    #pragma unroll
    for (int j = 0; j < 12; ++j) {
        float v = xr[lane + 32 * j];
        vals[j] = v; s += v; ss += v * v;
    }
    s = wave_sum(s); ss = wave_sum(ss);
    const float mu = s * (1.0f / DM);
    const float var = ss * (1.0f / DM) - mu * mu;
    const float r = rsqrtf(var + 1e-5f);
    unsigned short* orow = out + (size_t)row * DM;
    #pragma unroll
    for (int j = 0; j < 12; ++j) {
        int i = lane + 32 * j;
        orow[i] = f2bf((vals[j] - mu) * r * w[i] + b[i]);
    }
}

// ---------------------------------------------------------------------------
// Tiled bf16 WMMA GEMM: C(MxN) = A(MxK) * W(NxK)^T, f32 accumulate.
// BM=128 BN=128 BK=32, 256 threads (8 waves, 4x2), 32x64 per wave
// (2 A-frags x 4 B-frags -> 8 WMMAs per K-step).
// Tile feed: global_load_async_to_lds_b128, double-buffered LDS.
// EPI: 0 plain | 1 retain=exp(-softplus(acc+bias[n])) | 2 (1-aux)*acc | 3 aux+acc
// ---------------------------------------------------------------------------
#define LROW 40                    // padded row: 32 bf16 + 8 pad (80B stride)
#define TBUF (128 * LROW)          // one buffer, elements

template <int EPI>
__global__ __launch_bounds__(256)
void gemm_wmma_bf16(const unsigned short* __restrict__ A, int lda,
                    const unsigned short* __restrict__ Bw, int ldb,
                    float* __restrict__ C, int ldc,
                    const float* __restrict__ bias,
                    const float* __restrict__ aux,
                    int K) {
    __shared__ __align__(16) unsigned short Asm[2 * TBUF];
    __shared__ __align__(16) unsigned short Bsm[2 * TBUF];

    const int t = threadIdx.x;
    const int lane = t & 31, wid = t >> 5;
    const int wm = wid & 3, wn = wid >> 2;            // 4 x 2 wave grid
    const int row0 = blockIdx.y * 128, col0 = blockIdx.x * 128;

    v8f vz = {0.f, 0.f, 0.f, 0.f, 0.f, 0.f, 0.f, 0.f};
    v8f acc[2][4];
    #pragma unroll
    for (int mt = 0; mt < 2; ++mt)
        #pragma unroll
        for (int nt = 0; nt < 4; ++nt) acc[mt][nt] = vz;

    const unsigned aLds = (unsigned)(uintptr_t)&Asm[0];
    const unsigned bLds = (unsigned)(uintptr_t)&Bsm[0];

    // chunk map: 512 16B-chunks per tile, 2 per thread. c -> (row, sub)
    const int c0 = t, c1 = t + 256;
    const int ar0 = c0 >> 2, as0 = c0 & 3;
    const int ar1 = c1 >> 2, as1 = c1 & 3;
    const size_t aoff0 = (size_t)(row0 + ar0) * lda + as0 * 8;
    const size_t aoff1 = (size_t)(row0 + ar1) * lda + as1 * 8;
    const size_t boff0 = (size_t)(col0 + ar0) * ldb + as0 * 8;
    const size_t boff1 = (size_t)(col0 + ar1) * ldb + as1 * 8;
    const unsigned al0 = aLds + (unsigned)(ar0 * LROW + as0 * 8) * 2;
    const unsigned al1 = aLds + (unsigned)(ar1 * LROW + as1 * 8) * 2;
    const unsigned bl0 = bLds + (unsigned)(ar0 * LROW + as0 * 8) * 2;
    const unsigned bl1 = bLds + (unsigned)(ar1 * LROW + as1 * 8) * 2;

    const int KT = K >> 5;

    // prologue: feed tile 0 into buffer 0
    async_ld16(al0, A + aoff0);
    async_ld16(al1, A + aoff1);
    async_ld16(bl0, Bw + boff0);
    async_ld16(bl1, Bw + boff1);

    for (int kt = 0; kt < KT; ++kt) {
        wait_async0();          // this wave's tile-k async writes are in LDS
        __syncthreads();        // everyone's are

        if (kt + 1 < KT) {      // feed tile k+1 into the other buffer
            const int k1 = (kt + 1) * 32;
            const unsigned bsel = (unsigned)(((kt + 1) & 1) * TBUF * 2);
            async_ld16(al0 + bsel, A + aoff0 + k1);
            async_ld16(al1 + bsel, A + aoff1 + k1);
            async_ld16(bl0 + bsel, Bw + boff0 + k1);
            async_ld16(bl1 + bsel, Bw + boff1 + k1);
            if (kt + 2 < KT) {  // hint tile k+2 toward L2 (global_prefetch_b8)
                __builtin_prefetch(A + aoff0 + k1 + 32, 0, 1);
                __builtin_prefetch(Bw + boff0 + k1 + 32, 0, 1);
            }
        }

        const int base = (kt & 1) * TBUF;
        // A frag: lane holds row M=lane&15; K halves per ISA 16-bit A 16x32 layout
        FragBF fa[2], fb[4];
        const int kh = (lane >> 4) * 8;
        #pragma unroll
        for (int mt = 0; mt < 2; ++mt) {
            int r = wm * 32 + mt * 16 + (lane & 15);
            fa[mt].q[0] = *(const uint4*)&Asm[base + r * LROW + kh];
            fa[mt].q[1] = *(const uint4*)&Asm[base + r * LROW + 16 + kh];
        }
        // B frag: lane holds column n=lane&15, K halves per ISA 16-bit layout (same as A)
        const int koff = (lane >> 4) * 8;
        #pragma unroll
        for (int nt = 0; nt < 4; ++nt) {
            int n = wn * 64 + nt * 16 + (lane & 15);
            fb[nt].q[0] = *(const uint4*)&Bsm[base + n * LROW + koff];
            fb[nt].q[1] = *(const uint4*)&Bsm[base + n * LROW + koff + 16];
        }
        #pragma unroll
        for (int mt = 0; mt < 2; ++mt)
            #pragma unroll
            for (int nt = 0; nt < 4; ++nt)
                acc[mt][nt] = __builtin_amdgcn_wmma_f32_16x16x32_bf16(
                    false, fa[mt].v, false, fb[nt].v,
                    (short)0, acc[mt][nt], false, false);
        __syncthreads();        // done reading this buffer
    }

    // Epilogue: C/D layout: VGPR v -> M = (lane>=16 ? 8 : 0)+v, N = lane&15
    #pragma unroll
    for (int mt = 0; mt < 2; ++mt)
        #pragma unroll
        for (int nt = 0; nt < 4; ++nt)
            #pragma unroll
            for (int v = 0; v < 8; ++v) {
                int m = row0 + wm * 32 + mt * 16 + ((lane >> 4) * 8 + v);
                int n = col0 + wn * 64 + nt * 16 + (lane & 15);
                size_t idx = (size_t)m * ldc + n;
                float a = acc[mt][nt][v];
                if (EPI == 1) {
                    float tt = a + bias[n];
                    float sp = (tt > 20.f) ? tt : log1pf(expf(tt));
                    a = expf(-sp);
                } else if (EPI == 2) {
                    a = (1.0f - aux[idx]) * a;
                } else if (EPI == 3) {
                    a = aux[idx] + a;
                }
                C[idx] = a;
            }
}

// ---------------------------------------------------------------------------
// Depthwise 3x3 SAME conv on (B,Di,H,W) [stored as (B,N,2Di) u-half] + SiLU.
// ---------------------------------------------------------------------------
__global__ __launch_bounds__(256)
void conv_silu_kernel(const float* __restrict__ xz,
                      const float* __restrict__ cw, const float* __restrict__ cb,
                      float* __restrict__ u, unsigned short* __restrict__ u_bf) {
    int tid = blockIdx.x * blockDim.x + threadIdx.x;
    if (tid >= MROWS * DI) return;
    const int d = tid % DI;
    int rem = tid / DI;
    const int w = rem % WW; rem /= WW;
    const int h = rem % HH;
    const int b = rem / HH;
    float sum = cb[d];
    #pragma unroll
    for (int dh = -1; dh <= 1; ++dh) {
        int hh = h + dh;
        if (hh < 0 || hh >= HH) continue;
        #pragma unroll
        for (int dw = -1; dw <= 1; ++dw) {
            int ww2 = w + dw;
            if (ww2 < 0 || ww2 >= WW) continue;
            size_t p = ((size_t)(b * NN + hh * WW + ww2)) * (2 * DI) + d;
            sum += xz[p] * cw[d * 9 + (dh + 1) * 3 + (dw + 1)];
        }
    }
    float s = sum * sigmoidf_(sum);   // SiLU
    u[tid] = s;
    u_bf[tid] = f2bf(s);
}

// ---------------------------------------------------------------------------
// Row scans (routes 0,1): thread per (b,h,d); fwd writes partial, bwd adds.
// ---------------------------------------------------------------------------
__global__ __launch_bounds__(256)
void scan_rows_kernel(const float* __restrict__ retain,
                      const float* __restrict__ basev,
                      const float* __restrict__ rs,   // (4,Di)
                      float* __restrict__ part) {
    int tid = blockIdx.x * blockDim.x + threadIdx.x;
    if (tid >= BB * HH * DI) return;
    const int d = tid % DI;
    const int bh = tid / DI;
    const size_t base = (size_t)bh * WW * DI + d;
    const float r0 = rs[0 * DI + d], r1 = rs[1 * DI + d];
    float hst = 0.f;
    for (int w = 0; w < WW; ++w) {
        size_t p = base + (size_t)w * DI;
        hst = retain[p] * hst + basev[p] * r0;
        part[p] = hst;
    }
    hst = 0.f;
    for (int w = WW - 1; w >= 0; --w) {
        size_t p = base + (size_t)w * DI;
        hst = retain[p] * hst + basev[p] * r1;
        part[p] += hst;
    }
}

// ---------------------------------------------------------------------------
// Column scans (routes 2,3): thread per (b,w,d); fwd adds, bwd merges *0.25.
// ---------------------------------------------------------------------------
__global__ __launch_bounds__(256)
void scan_cols_kernel(const float* __restrict__ retain,
                      const float* __restrict__ basev,
                      const float* __restrict__ rs,
                      float* __restrict__ part) {
    int tid = blockIdx.x * blockDim.x + threadIdx.x;
    if (tid >= BB * WW * DI) return;
    const int d = tid % DI;
    const int bw = tid / DI;
    const int b = bw / WW, w = bw % WW;
    const size_t base = ((size_t)b * NN + w) * DI + d;
    const size_t step = (size_t)WW * DI;
    const float r2 = rs[2 * DI + d], r3 = rs[3 * DI + d];
    float hst = 0.f;
    for (int h = 0; h < HH; ++h) {
        size_t p = base + (size_t)h * step;
        hst = retain[p] * hst + basev[p] * r2;
        part[p] += hst;
    }
    hst = 0.f;
    for (int h = HH - 1; h >= 0; --h) {
        size_t p = base + (size_t)h * step;
        hst = retain[p] * hst + basev[p] * r3;
        part[p] = 0.25f * (part[p] + hst);   // final merged
    }
}

// ---------------------------------------------------------------------------
// out-LN(merged) + u*D_param, then gate with SiLU(z); bf16 output for GEMM.
// ---------------------------------------------------------------------------
__global__ __launch_bounds__(256)
void gate_kernel(const float* __restrict__ merged, const float* __restrict__ u,
                 const float* __restrict__ xz,
                 const float* __restrict__ onw, const float* __restrict__ onb,
                 const float* __restrict__ Dp,
                 unsigned short* __restrict__ gate_bf) {
    const int lane = threadIdx.x & 31, wid = threadIdx.x >> 5;
    const int row = blockIdx.x * 8 + wid;
    const float* mr = merged + (size_t)row * DI;
    float vals[24], s = 0.f, ss = 0.f;
    #pragma unroll
    for (int j = 0; j < 24; ++j) {
        float v = mr[lane + 32 * j];
        vals[j] = v; s += v; ss += v * v;
    }
    s = wave_sum(s); ss = wave_sum(ss);
    const float mu = s * (1.0f / DI);
    const float var = ss * (1.0f / DI) - mu * mu;
    const float r = rsqrtf(var + 1e-5f);
    const float* ur = u + (size_t)row * DI;
    const float* zr = xz + (size_t)row * (2 * DI) + DI;
    unsigned short* gr = gate_bf + (size_t)row * DI;
    #pragma unroll
    for (int j = 0; j < 24; ++j) {
        int i = lane + 32 * j;
        float y = (vals[j] - mu) * r * onw[i] + onb[i] + ur[i] * Dp[i];
        float z = zr[i];
        gr[i] = f2bf(y * z * sigmoidf_(z));
    }
}

// ---------------------------------------------------------------------------
// Host-side orchestration
// ---------------------------------------------------------------------------
extern "C" void kernel_launch(void* const* d_in, const int* in_sizes, int n_in,
                              void* d_out, int out_size, void* d_ws, size_t ws_size,
                              hipStream_t stream) {
    const float* x        = (const float*)d_in[0];
    const float* norm_w   = (const float*)d_in[1];
    const float* norm_b   = (const float*)d_in[2];
    const float* inproj_w = (const float*)d_in[3];
    const float* conv_w   = (const float*)d_in[4];
    const float* conv_b   = (const float*)d_in[5];
    const float* delta_w  = (const float*)d_in[6];
    const float* delta_b  = (const float*)d_in[7];
    const float* value_w  = (const float*)d_in[8];
    const float* route_sc = (const float*)d_in[9];
    const float* onorm_w  = (const float*)d_in[10];
    const float* onorm_b  = (const float*)d_in[11];
    const float* D_param  = (const float*)d_in[12];
    const float* outproj_w= (const float*)d_in[13];

    char* ws = (char*)d_ws;
    size_t off = 0;
    auto alloc = [&](size_t bytes) { char* p = ws + off; off += (bytes + 255) & ~(size_t)255; return p; };

    unsigned short* wbf_inproj  = (unsigned short*)alloc((size_t)2 * DI * DM * 2);
    unsigned short* wbf_delta   = (unsigned short*)alloc((size_t)DI * DI * 2);
    unsigned short* wbf_value   = (unsigned short*)alloc((size_t)DI * DI * 2);
    unsigned short* wbf_outproj = (unsigned short*)alloc((size_t)DM * DI * 2);
    unsigned short* xln_bf      = (unsigned short*)alloc((size_t)MROWS * DM * 2);
    float*          xz          = (float*)alloc((size_t)MROWS * 2 * DI * 4);
    float*          u           = (float*)alloc((size_t)MROWS * DI * 4);
    unsigned short* u_bf        = (unsigned short*)alloc((size_t)MROWS * DI * 2);
    float*          retain      = (float*)alloc((size_t)MROWS * DI * 4);
    float*          basev       = (float*)alloc((size_t)MROWS * DI * 4);
    float*          part        = (float*)alloc((size_t)MROWS * DI * 4);
    unsigned short* gate_bf     = (unsigned short*)alloc((size_t)MROWS * DI * 2);
    (void)ws_size; (void)n_in; (void)in_sizes; (void)out_size;

    // 1) weight conversions
    cvt_bf16_kernel<<<1152, 256, 0, stream>>>(inproj_w,  wbf_inproj,  2 * DI * DM);
    cvt_bf16_kernel<<<1152, 256, 0, stream>>>(delta_w,   wbf_delta,   DI * DI);
    cvt_bf16_kernel<<<1152, 256, 0, stream>>>(value_w,   wbf_value,   DI * DI);
    cvt_bf16_kernel<<<576,  256, 0, stream>>>(outproj_w, wbf_outproj, DM * DI);

    // 2) LayerNorm -> bf16
    ln_in_kernel<<<MROWS / 8, 256, 0, stream>>>(x, norm_w, norm_b, xln_bf);

    // 3) in_proj GEMM: (32768x384)x(384x1536) -> xz
    gemm_wmma_bf16<0><<<dim3((2 * DI) / 128, MROWS / 128), 256, 0, stream>>>(
        xln_bf, DM, wbf_inproj, DM, xz, 2 * DI, nullptr, nullptr, DM);

    // 4) depthwise conv + SiLU
    conv_silu_kernel<<<(MROWS * DI) / 256, 256, 0, stream>>>(xz, conv_w, conv_b, u, u_bf);

    // 5) delta GEMM with retain epilogue
    gemm_wmma_bf16<1><<<dim3(DI / 128, MROWS / 128), 256, 0, stream>>>(
        u_bf, DI, wbf_delta, DI, retain, DI, delta_b, nullptr, DI);

    // 6) value GEMM with (1-retain)*v epilogue
    gemm_wmma_bf16<2><<<dim3(DI / 128, MROWS / 128), 256, 0, stream>>>(
        u_bf, DI, wbf_value, DI, basev, DI, nullptr, retain, DI);

    // 7) 4 directional scans
    scan_rows_kernel<<<(BB * HH * DI) / 256, 256, 0, stream>>>(retain, basev, route_sc, part);
    scan_cols_kernel<<<(BB * WW * DI) / 256, 256, 0, stream>>>(retain, basev, route_sc, part);

    // 8) out-LN + D skip + SiLU gate
    gate_kernel<<<MROWS / 8, 256, 0, stream>>>(part, u, xz, onorm_w, onorm_b, D_param, gate_bf);

    // 9) out_proj GEMM with residual-add epilogue -> d_out
    gemm_wmma_bf16<3><<<dim3(DM / 128, MROWS / 128), 256, 0, stream>>>(
        gate_bf, DI, wbf_outproj, DI, (float*)d_out, DM, nullptr, x, DI);
}
